// RNNEncoder_82935818486036
// MI455X (gfx1250) — matrix operand order repaired and checked by reading, not compiled, not run on hardware
//
#include <hip/hip_runtime.h>
#include <hip/hip_bf16.h>

// ---------------- problem constants ----------------
#define BB 64
#define TT 512
#define DIN 768
#define HH 512
#define G4 2048   // 4*H
#define HN 1024   // 2*H
#define NFC1 512
#define NFC2 256
#define LN_EPS 1e-3f

typedef __attribute__((ext_vector_type(16))) _Float16 v16h;
typedef __attribute__((ext_vector_type(4)))  _Float16 v4h;
typedef __attribute__((ext_vector_type(8)))  float    v8f;
typedef __attribute__((ext_vector_type(4)))  float    f32x4;
typedef __attribute__((ext_vector_type(4)))  unsigned int u32x4;
typedef __attribute__((ext_vector_type(8)))  int      i32x8;
typedef __attribute__((ext_vector_type(4)))  int      i32x4;

__device__ __forceinline__ float sigmoidf_(float x) { return 1.f / (1.f + __expf(-x)); }

// block-wide reductions over 256 threads (8 wave32s) via LDS
__device__ __forceinline__ float blk_sum(float v, float* red) {
  int tid = threadIdx.x;
  red[tid] = v; __syncthreads();
  for (int off = 128; off > 0; off >>= 1) {
    if (tid < off) red[tid] += red[tid + off];
    __syncthreads();
  }
  float r = red[0]; __syncthreads();
  return r;
}
__device__ __forceinline__ float blk_max(float v, float* red) {
  int tid = threadIdx.x;
  red[tid] = v; __syncthreads();
  for (int off = 128; off > 0; off >>= 1) {
    if (tid < off) red[tid] = fmaxf(red[tid], red[tid + off]);
    __syncthreads();
  }
  float r = red[0]; __syncthreads();
  return r;
}

// ---------------------------------------------------------------------------
// TDM: async-load a [64 x 128-half] f16 tile (global row stride 512 halves)
// into LDS with hardware padding of 4 dwords after every 64 dwords, i.e. an
// LDS row stride of 136 halves (bank-conflict-free A-fragment reads).
// D# layout per CDNA5 ISA.
// ---------------------------------------------------------------------------
__device__ __forceinline__ void tdm_load_h_chunk(const _Float16* gsrc,
                                                 unsigned lds_byte_off) {
  unsigned long long ga = (unsigned long long)(uintptr_t)gsrc;
  u32x4 g0;
  g0[0] = 1u;                                   // count=1, user descriptor
  g0[1] = lds_byte_off;                         // lds_addr
  g0[2] = (unsigned)ga;                         // global_addr[31:0]
  g0[3] = (unsigned)((ga >> 32) & 0x1FFFFFFu)   // global_addr[56:32]
          | (2u << 30);                         // type=2 ("image")
  i32x8 g1;
  g1[0] = (int)((1u << 16)      // data_size = 1 -> 2 bytes
              | (1u << 20)      // pad_enable
              | (5u << 22)      // pad_interval = 64 dwords (tile row)
              | (3u << 25));    // pad_amount   = 4 dwords (8 halves)
  g1[1] = (int)(512u << 16);    // tensor_dim0 = 512 (bits 79:48, low half)
  g1[2] = (int)(64u << 16);     // tensor_dim0 hi=0 | tensor_dim1 = 64
  g1[3] = (int)(128u << 16);    // tensor_dim1 hi=0 | tile_dim0 = 128
  g1[4] = 64;                   // tile_dim1 = 64, tile_dim2 = 0
  g1[5] = 512;                  // tensor_dim0_stride = 512 (48b, low word)
  g1[6] = 0;
  g1[7] = 0;                    // tensor_dim1_stride = 0 (2D)
  i32x4 z4 = {0, 0, 0, 0};
#if __clang_major__ >= 23
  i32x8 z8 = {0, 0, 0, 0, 0, 0, 0, 0};
  __builtin_amdgcn_tensor_load_to_lds(g0, g1, z4, z4, z8, 0);
#else
  __builtin_amdgcn_tensor_load_to_lds(g0, g1, z4, z4, 0);
#endif
}

// ---------------------------------------------------------------------------
// Kernel 1: convert U_f / U_b (f32) -> f16 once; 4 MB total, stays in L2.
// ---------------------------------------------------------------------------
__global__ void cvt_u_kernel(const float* __restrict__ Uf,
                             const float* __restrict__ Ub,
                             _Float16* __restrict__ U16) {
  const int N1 = HH * G4;                // 1,048,576 per direction
  int i4 = (blockIdx.x * blockDim.x + threadIdx.x) * 4;   // grid covers 2*N1/4
  const float* src = (i4 < N1) ? (Uf + i4) : (Ub + (i4 - N1));
  f32x4 f = *(const f32x4*)src;
  v4h h = { (_Float16)f[0], (_Float16)f[1], (_Float16)f[2], (_Float16)f[3] };
  *(v4h*)&U16[i4] = h;
}

// ---------------------------------------------------------------------------
// Kernel 2: zero h/c state (fresh every call; ws is poisoned by the harness)
// ---------------------------------------------------------------------------
__global__ void init_state_kernel(float* __restrict__ h32,
                                  _Float16* __restrict__ h16,  // [2 bufs][2 dirs][B][H]
                                  float* __restrict__ c32) {
  const int NS = 2 * BB * HH;            // per-array f32 state count
  int i = blockIdx.x * blockDim.x + threadIdx.x;   // grid covers 2*NS
  if (i < NS) { h32[i] = 0.f; c32[i] = 0.f; }
  h16[i] = (_Float16)0.f;                // both double-buffer halves
}

// ---------------------------------------------------------------------------
// Kernel 3: gx[dir] = x @ W_dir + b_dir   (M=B*T=32768, K=768, N=2048), f16 out
// Workgroup: 256 thr (8 waves) -> 64x64 output tile; wave = (m_tile, n_half),
// two v_wmma_f32_16x16x32_f16 per 32-wide K-step. Staging is vectorized:
// float4 global loads -> f16x4 LDS stores.
// ---------------------------------------------------------------------------
#define LDA (32 + 8)   // x LDS tile row stride (halves) — break bank conflicts
#define LDB (64 + 8)   // W LDS tile row stride

__global__ void gx_gemm_kernel(const float* __restrict__ x,
                               const float* __restrict__ Wf,
                               const float* __restrict__ Wb,
                               const float* __restrict__ bf,
                               const float* __restrict__ bb,
                               _Float16* __restrict__ gx) {
  __shared__ _Float16 xs[64 * LDA];
  __shared__ _Float16 wsl[32 * LDB];

  const int dir = blockIdx.z;
  const int m0  = blockIdx.y * 64;          // row block in [0, 32768)
  const int n0  = blockIdx.x * 64;          // col block in [0, 2048)
  const float* W    = dir ? Wb : Wf;
  const float* bias = dir ? bb : bf;

  const int tid  = threadIdx.x;
  const int wave = tid >> 5;
  const int lane = tid & 31;
  const int wm   = wave >> 1;               // 0..3 : 16-row tile
  const int wn   = wave & 1;                // 0..1 : 32-col half

  v8f acc0 = {}; v8f acc1 = {};

  for (int k0 = 0; k0 < DIN; k0 += 32) {
    // stage x tile [64][32]: 512 float4 vectors, 2 per thread
    #pragma unroll
    for (int it = 0; it < 2; ++it) {
      const int idx = tid + it * 256;
      const int r = idx >> 3, c4 = (idx & 7) * 4;
      f32x4 f = *(const f32x4*)&x[(size_t)(m0 + r) * DIN + k0 + c4];
      v4h h = { (_Float16)f[0], (_Float16)f[1], (_Float16)f[2], (_Float16)f[3] };
      *(v4h*)&xs[r * LDA + c4] = h;
    }
    // stage W tile [32][64]: 512 float4 vectors, 2 per thread
    #pragma unroll
    for (int it = 0; it < 2; ++it) {
      const int idx = tid + it * 256;
      const int r = idx >> 4, c4 = (idx & 15) * 4;
      f32x4 f = *(const f32x4*)&W[(size_t)(k0 + r) * G4 + n0 + c4];
      v4h h = { (_Float16)f[0], (_Float16)f[1], (_Float16)f[2], (_Float16)f[3] };
      *(v4h*)&wsl[r * LDB + c4] = h;
    }
    __syncthreads();

    // A fragment (16x32 f16): lane L<16 -> M=L, K halves {kb..kb+7, 16+kb..};
    // kb = 0 for lanes 0-15, 8 for lanes 16-31 (ISA 7.12.2 table)
    v16h afrag;
    {
      const _Float16* ap = &xs[(wm * 16 + (lane & 15)) * LDA];
      const int kb = (lane >> 4) * 8;
      #pragma unroll
      for (int j = 0; j < 8; ++j) {
        afrag[j]     = ap[kb + j];
        afrag[8 + j] = ap[16 + kb + j];
      }
    }
    // B fragments (32x16 f16): lane l holds column N=l%16, K halves {kb..kb+7, 16+kb..}
    v16h b0, b1;
    {
      const _Float16* bp = &wsl[wn * 32 + (lane & 15)];
      const int kb = (lane >> 4) * 8;
      #pragma unroll
      for (int j = 0; j < 8; ++j) {
        b0[j]     = bp[(kb + j) * LDB];
        b0[8 + j] = bp[(16 + kb + j) * LDB];
        b1[j]     = bp[(kb + j) * LDB + 16];
        b1[8 + j] = bp[(16 + kb + j) * LDB + 16];
      }
    }
    acc0 = __builtin_amdgcn_wmma_f32_16x16x32_f16(false, afrag, false, b0,
                                                  (short)0, acc0, false, false);
    acc1 = __builtin_amdgcn_wmma_f32_16x16x32_f16(false, afrag, false, b1,
                                                  (short)0, acc1, false, false);
    __syncthreads();
  }

  // C/D layout: VGPR r, lanes<16 -> M=r, lanes>=16 -> M=r+8; N = lane%16
  const int mrow  = m0 + wm * 16 + (lane >> 4) * 8;
  const int ncol0 = n0 + wn * 32 + (lane & 15);
  const float bias0 = bias[ncol0];
  const float bias1 = bias[ncol0 + 16];
  const size_t base = (size_t)dir * (size_t)(BB * TT) * G4;
  #pragma unroll
  for (int r = 0; r < 8; ++r) {
    size_t row = base + (size_t)(mrow + r) * G4;
    gx[row + ncol0]      = (_Float16)(acc0[r] + bias0);
    gx[row + ncol0 + 16] = (_Float16)(acc1[r] + bias1);
  }
}

// ---------------------------------------------------------------------------
// Kernel 4: one LSTM time step, fwd+bwd fused (grid.y = dir).
// Workgroup owns 32 hidden units (jtile) and computes ALL FOUR gates for all
// 64 batch rows via WMMA (wave = gate x j-half), then finishes the cell update
// in-kernel through an LDS gate exchange. h-state f16 is double-buffered
// across steps (read buf t&1, write buf (t+1)&1) to avoid inter-WG races.
// The h tile is staged into LDS by the Tensor Data Mover (async DMA with
// hardware padding), fenced by s_wait_tensorcnt + workgroup barrier.
// ---------------------------------------------------------------------------
#define LDH (128 + 8)   // h LDS chunk row stride (halves) == TDM pad layout

__global__ void lstm_step_kernel(const _Float16* __restrict__ U16,  // [2][512][2048]
                                 const _Float16* __restrict__ gx,   // [2][B*T][2048]
                                 const unsigned char* __restrict__ mask, // [B][T] bool
                                 float* __restrict__ h32,           // [2][B][H]
                                 _Float16* __restrict__ h16,        // [2][2][B][H]
                                 float* __restrict__ c32,           // [2][B][H]
                                 float* __restrict__ hn,            // d_out [B][T][2H]
                                 int t) {
  __shared__ _Float16 hs[64 * LDH];
  __shared__ float zb[4][64][34];     // [gate][batch][j-local], padded

  const int dir   = blockIdx.y;
  const int jt    = blockIdx.x;       // 0..15 -> 32 hidden units
  const int t_eff = dir ? (TT - 1 - t) : t;
  const int tid   = threadIdx.x;
  const int wave  = tid >> 5;
  const int lane  = tid & 31;
  const int g     = wave >> 1;        // gate: 0=i 1=f 2=cc 3=o
  const int jh    = wave & 1;         // j half within the 32-wide slice
  const int ncol  = g * HH + jt * 32 + jh * 16;   // column base in U / z

  const _Float16* U    = U16 + (size_t)dir * HH * G4;
  const _Float16* hsrc = h16 + ((size_t)(t & 1) * 2 + dir) * BB * HH;
  _Float16*       hdst = h16 + ((size_t)((t + 1) & 1) * 2 + dir) * BB * HH;
  const unsigned  hs_off = (unsigned)(unsigned long long)(uintptr_t)hs;

  // warm the gx row(s) this block's pointwise phase will read
  {
    int pb = tid >> 2, seg = tid & 3;
    size_t grow = ((size_t)dir * (BB * TT) + (size_t)pb * TT + t_eff) * G4;
    __builtin_prefetch(&gx[grow + seg * HH + jt * 32], 0, 1);
  }

  v8f acc[4] = { {}, {}, {}, {} };

  for (int kb = 0; kb < HH; kb += 128) {
    // async DMA of h chunk [64 batches][128 K] into padded LDS rows
    if (tid == 0) {
      tdm_load_h_chunk(hsrc + kb, hs_off);
      __builtin_amdgcn_s_wait_tensorcnt(0);
    }
    __syncthreads();

    for (int ki = 0; ki < 128; ki += 32) {
      // B fragment straight from (L2-resident) f16 U: lane = N column, K in table order
      v16h bfrag;
      {
        const _Float16* bq = &U[(size_t)(kb + ki) * G4 + ncol + (lane & 15)];
        const int kq = (lane >> 4) * 8;
        #pragma unroll
        for (int j = 0; j < 8; ++j) {
          bfrag[j]     = bq[(size_t)(kq + j) * G4];
          bfrag[8 + j] = bq[(size_t)(16 + kq + j) * G4];
        }
      }
      #pragma unroll
      for (int mt = 0; mt < 4; ++mt) {          // batch tiles 0..63
        v16h afrag;
        const _Float16* ap = &hs[(mt * 16 + (lane & 15)) * LDH + ki];
        const int kbse = (lane >> 4) * 8;
        #pragma unroll
        for (int j = 0; j < 8; ++j) {
          afrag[j]     = ap[kbse + j];
          afrag[8 + j] = ap[16 + kbse + j];
        }
        acc[mt] = __builtin_amdgcn_wmma_f32_16x16x32_f16(false, afrag, false, bfrag,
                                                         (short)0, acc[mt], false, false);
      }
    }
    __syncthreads();
  }

  // scatter this wave's gate values into the LDS exchange buffer
  {
    const int rofs = (lane >> 4) * 8;
    const int n    = lane & 15;
    #pragma unroll
    for (int mt = 0; mt < 4; ++mt)
      #pragma unroll
      for (int r = 0; r < 8; ++r)
        zb[g][mt * 16 + rofs + r][jh * 16 + n] = acc[mt][r];
  }
  __syncthreads();

  // pointwise cell update for (64 batches x 32 hidden units)
  #pragma unroll
  for (int it = 0; it < 8; ++it) {
    const int i  = tid + it * 256;
    const int b  = i >> 5;
    const int jl = i & 31;
    const int j  = jt * 32 + jl;
    const size_t grow = ((size_t)dir * (BB * TT) + (size_t)b * TT + t_eff) * G4;
    float zi = zb[0][b][jl] + (float)gx[grow + 0 * HH + j];
    float zf = zb[1][b][jl] + (float)gx[grow + 1 * HH + j];
    float zc = zb[2][b][jl] + (float)gx[grow + 2 * HH + j];
    float zo = zb[3][b][jl] + (float)gx[grow + 3 * HH + j];

    const size_t sidx = (size_t)dir * BB * HH + (size_t)b * HH + j;
    const float cp = c32[sidx];
    const float hp = h32[sidx];
    const float cn = sigmoidf_(zf) * cp + sigmoidf_(zi) * tanhf(zc);
    const float hv = sigmoidf_(zo) * tanhf(cn);
    const bool  m  = mask[b * TT + t_eff] != 0;
    const float h2 = m ? hv : hp;
    const float c2 = m ? cn : cp;

    c32[sidx] = c2;
    h32[sidx] = h2;
    hdst[(size_t)b * HH + j] = (_Float16)h2;
    hn[((size_t)b * TT + t_eff) * HN + dir * HH + j] = h2;   // pre-LN scratch
  }
}

// ---------------------------------------------------------------------------
// Kernel 5: fused LayerNorm (in place on d_out's hn region) + attention score
// ---------------------------------------------------------------------------
__global__ void ln_score_kernel(float* __restrict__ hn,
                                const float* __restrict__ gamma,
                                const float* __restrict__ beta,
                                const float* __restrict__ attn_w,
                                const float* __restrict__ attn_b,
                                float* __restrict__ scores) {
  __shared__ float red[256];
  const int row = blockIdx.x;                 // b*T + t
  float* p = hn + (size_t)row * HN;
  const int tid = threadIdx.x;

  float v[4], s = 0.f;
  #pragma unroll
  for (int j = 0; j < 4; ++j) { v[j] = p[tid + 256 * j]; s += v[j]; }
  const float mu = blk_sum(s, red) * (1.f / HN);

  float sq = 0.f;
  #pragma unroll
  for (int j = 0; j < 4; ++j) { float d = v[j] - mu; sq += d * d; }
  const float rstd = rsqrtf(blk_sum(sq, red) * (1.f / HN) + LN_EPS);

  float sc = 0.f;
  #pragma unroll
  for (int j = 0; j < 4; ++j) {
    const int d = tid + 256 * j;
    const float y = (v[j] - mu) * rstd * gamma[d] + beta[d];
    p[d] = y;
    sc += y * attn_w[d];
  }
  const float tot = blk_sum(sc, red);
  if (tid == 0) scores[row] = tot + attn_b[0];
}

// ---------------------------------------------------------------------------
// Kernel 6: per-batch softmax over T + weighted context reduction
// ---------------------------------------------------------------------------
__global__ void softmax_ctx_kernel(const float* __restrict__ hn,
                                   const float* __restrict__ scores,
                                   float* __restrict__ ctx) {
  __shared__ float w[TT];
  __shared__ float red[256];
  const int b = blockIdx.x, tid = threadIdx.x;

  float m = -1e30f;
  for (int t = tid; t < TT; t += 256) m = fmaxf(m, scores[b * TT + t]);
  const float mx = blk_max(m, red);

  float s = 0.f;
  for (int t = tid; t < TT; t += 256) {
    float e = __expf(scores[b * TT + t] - mx);
    w[t] = e; s += e;
  }
  const float inv = 1.f / blk_sum(s, red);
  for (int t = tid; t < TT; t += 256) w[t] *= inv;
  __syncthreads();

  for (int d = tid; d < HN; d += 256) {
    const float* hp = hn + (size_t)b * TT * HN + d;
    float a = 0.f;
    for (int t = 0; t < TT; ++t) a += w[t] * hp[(size_t)t * HN];
    ctx[b * HN + d] = a;
  }
}

// ---------------------------------------------------------------------------
// Kernels 7/8: small GEMV FCs (relu / tanh)
// ---------------------------------------------------------------------------
__global__ void fc1_kernel(const float* __restrict__ ctx,
                           const float* __restrict__ w,
                           const float* __restrict__ bias,
                           float* __restrict__ out) {
  const int idx = blockIdx.x * 256 + threadIdx.x;   // 64*512
  const int b = idx >> 9, n = idx & 511;
  const float* c = ctx + (size_t)b * HN;
  float a = bias[n];
  for (int k = 0; k < HN; ++k) a += c[k] * w[(size_t)k * NFC1 + n];
  out[idx] = fmaxf(a, 0.f);
}
__global__ void fc2_kernel(const float* __restrict__ in,
                           const float* __restrict__ w,
                           const float* __restrict__ bias,
                           float* __restrict__ out) {
  const int idx = blockIdx.x * 256 + threadIdx.x;   // 64*256
  const int b = idx >> 8, n = idx & 255;
  const float* c = in + (size_t)b * NFC1;
  float a = bias[n];
  for (int k = 0; k < NFC1; ++k) a += c[k] * w[(size_t)k * NFC2 + n];
  out[idx] = tanhf(a);
}

// ---------------------------------------------------------------------------
extern "C" void kernel_launch(void* const* d_in, const int* in_sizes, int n_in,
                              void* d_out, int out_size, void* d_ws, size_t ws_size,
                              hipStream_t stream) {
  (void)in_sizes; (void)n_in; (void)out_size; (void)ws_size;

  const float*         x     = (const float*)d_in[0];
  const unsigned char* mask  = (const unsigned char*)d_in[1];   // jnp bool -> 1B
  const float*         W_f   = (const float*)d_in[2];
  const float*         U_f   = (const float*)d_in[3];
  const float*         b_f   = (const float*)d_in[4];
  const float*         W_b   = (const float*)d_in[5];
  const float*         U_b   = (const float*)d_in[6];
  const float*         b_b   = (const float*)d_in[7];
  const float*         lng   = (const float*)d_in[8];
  const float*         lnb   = (const float*)d_in[9];
  const float*         attnw = (const float*)d_in[10];
  const float*         attnb = (const float*)d_in[11];
  const float*         fc1w  = (const float*)d_in[12];
  const float*         fc1b  = (const float*)d_in[13];
  const float*         fc2w  = (const float*)d_in[14];
  const float*         fc2b  = (const float*)d_in[15];

  float* hn      = (float*)d_out;                       // [B][T][2H]
  float* out_ctx = hn + (size_t)BB * TT * HN;           // [B][FC2]

  // ---- workspace carve-up ----
  char* wsb = (char*)d_ws;
  size_t off = 0;
  auto carve = [&](size_t bytes) -> void* {
    void* p = wsb + off;
    off = (off + bytes + 255) & ~(size_t)255;
    return p;
  };
  _Float16* U16 = (_Float16*)carve((size_t)2 * HH * G4 * sizeof(_Float16));        // 4 MB
  _Float16* GX  = (_Float16*)carve((size_t)2 * BB * TT * G4 * sizeof(_Float16));   // 256 MB
  float*    H32 = (float*)   carve((size_t)2 * BB * HH * sizeof(float));
  _Float16* H16 = (_Float16*)carve((size_t)2 * 2 * BB * HH * sizeof(_Float16));    // dbl-buf
  float*    C32 = (float*)   carve((size_t)2 * BB * HH * sizeof(float));
  float*    SC  = (float*)   carve((size_t)BB * TT * sizeof(float));
  float*    CTX = (float*)   carve((size_t)BB * HN * sizeof(float));
  float*    F1  = (float*)   carve((size_t)BB * NFC1 * sizeof(float));

  // U conversion: 2*1,048,576 elems / 4 per thread / 256 per block = 2048 blocks
  hipLaunchKernelGGL(cvt_u_kernel, dim3(2048), dim3(256), 0, stream, U_f, U_b, U16);
  // state init: 2*65536 threads
  hipLaunchKernelGGL(init_state_kernel, dim3(512), dim3(256), 0, stream, H32, H16, C32);

  // big input-projection GEMM: 32768x768 @ 768x2048, both directions
  hipLaunchKernelGGL(gx_gemm_kernel, dim3(G4 / 64, (BB * TT) / 64, 2), dim3(256),
                     0, stream, x, W_f, W_b, b_f, b_b, GX);

  // sequential scan: one launch per step, fwd+bwd fused
  for (int t = 0; t < TT; ++t)
    hipLaunchKernelGGL(lstm_step_kernel, dim3(HH / 32, 2), dim3(256), 0, stream,
                       U16, GX, mask, H32, H16, C32, hn, t);

  hipLaunchKernelGGL(ln_score_kernel, dim3(BB * TT), dim3(256), 0, stream,
                     hn, lng, lnb, attnw, attnb, SC);
  hipLaunchKernelGGL(softmax_ctx_kernel, dim3(BB), dim3(256), 0, stream, hn, SC, CTX);
  hipLaunchKernelGGL(fc1_kernel, dim3(BB * NFC1 / 256), dim3(256), 0, stream,
                     CTX, fc1w, fc1b, F1);
  hipLaunchKernelGGL(fc2_kernel, dim3(BB * NFC2 / 256), dim3(256), 0, stream,
                     F1, fc2w, fc2b, out_ctx);
}
